// RevGRU_52621939310804
// MI455X (gfx1250) — compile-verified, matrix operand order checked by reading, not run on hardware
//
#include <hip/hip_runtime.h>

// ---------------------------------------------------------------------------
// RevGRU step for MI455X (gfx1250, wave32, WMMA).
// 4 serial GEMMs on the bf16 WMMA pipe (v_wmma_f32_16x16x32_bf16).
// v2: pre-transposed bf16 weights (no in-loop LDS transpose), 128x128 block
// tile / 32x64 wave tile (8 WMMA per wave per K-step), register
// double-buffered global->LDS staging.
// ---------------------------------------------------------------------------

#define BATCH   8192
#define IN_SZ   1024
#define HSZ     2048
#define GSZ     1024
#define KDIM    2048        // IN_SZ + GSZ
#define MAXF    0.875f
#define INV2P23 (1.0f / 8388608.0f)
#define F2P23   8388608.0f

typedef __bf16 bf16_t;
typedef __attribute__((ext_vector_type(16))) __bf16 v16bf;
typedef __attribute__((ext_vector_type(8)))  __bf16 v8bf;
typedef __attribute__((ext_vector_type(8)))  float  v8f;

__device__ __forceinline__ float sigmoid_(float x) {
    return 1.0f / (1.0f + expf(-x));
}

// ---------------------------------------------------------------------------
// f32 W[K][N] -> bf16 Wt[N][K] (transpose + convert, one time per launch).
// Coalesced reads; scattered 2B writes are fine for a 25 MB one-shot.
// nbits = log2(N).
// ---------------------------------------------------------------------------
__global__ void cvtT_kernel(const float* __restrict__ W,
                            bf16_t* __restrict__ Wt,
                            int Kd, int nbits) {
    int i = blockIdx.x * blockDim.x + threadIdx.x;   // over Kd * Nd
    int k = i >> nbits;
    int n = i & ((1 << nbits) - 1);
    Wt[(size_t)n * Kd + k] = (bf16_t)W[i];
}

// ---------------------------------------------------------------------------
// Build A = [x | h2_fl] in bf16; zero the optimal_bits accumulator slot.
// ---------------------------------------------------------------------------
__global__ void prep_kernel(const float* __restrict__ x,
                            const int* __restrict__ hidden,
                            bf16_t* __restrict__ A,
                            float* __restrict__ bits) {
    int i = blockIdx.x * blockDim.x + threadIdx.x;
    if (i == 0) *bits = 0.0f;
    int row = i >> 11;            // / HSZ
    int col = i & (HSZ - 1);
    float v;
    if (col < IN_SZ) v = x[row * IN_SZ + col];
    else             v = (float)hidden[row * HSZ + col] * INV2P23;  // h2 half
    A[i] = (bf16_t)v;
}

// ---------------------------------------------------------------------------
// bf16 WMMA GEMM:  Out[M,N] = A[M,K] @ Wt[N,K]^T + bias
//   A  : row-major [M][K] bf16
//   Wt : row-major [N][K] bf16 (pre-transposed weights)
// Block: 256 threads = 8 wave32. Block tile 128(M) x 128(N), K-step 32.
// Wave tile 32x64 -> 2x4 WMMA tiles -> 8 v_wmma per wave per K-step.
// Both LDS tiles stored K-contiguous (padded stride 40 bf16 = 80B), so every
// fragment is two 16B ds_load_b128 per the ISA 7.12.2 16-bit A/B layout
// (lanes 0-15: K 0-7 / 16-23, lanes 16-31: K 8-15 / 24-31).
// Register double-buffering: next K-step's global loads issue before compute.
// ---------------------------------------------------------------------------
#define LPAD 40   // LDS row stride in bf16 elements (80B, 16B-aligned)

__global__ __launch_bounds__(256)
void gemm_bf16_kernel(const bf16_t* __restrict__ A,
                      const bf16_t* __restrict__ Wt,
                      const float* __restrict__ bias,
                      float* __restrict__ Out,
                      int M, int N, int K) {
    __shared__ bf16_t lA[128 * LPAD];
    __shared__ bf16_t lB[128 * LPAD];

    const int tid  = threadIdx.x;
    const int lane = tid & 31;
    const int wave = tid >> 5;
    const int bm = blockIdx.y * 128;
    const int bn = blockIdx.x * 128;
    const int wm = (wave & 3) * 32;   // wave row offset in block tile
    const int wn = (wave >> 2) * 64;  // wave col offset in block tile

    const int fr    = lane & 15;          // fragment row/col within 16
    const int khalf = (lane >> 4) << 3;   // 0 or 8 (K half select)

    // staging: each thread owns one 16-bf16 (32B) chunk of each tile
    const int srow = tid >> 1;            // 0..127
    const int skc  = (tid & 1) << 4;      // 0 or 16
    const bf16_t* ga = A  + (size_t)(bm + srow) * K + skc;
    const bf16_t* gb = Wt + (size_t)(bn + srow) * K + skc;

    // prologue: load K-step 0 into registers
    v8bf ra0 = *(const v8bf*)(ga);
    v8bf ra1 = *(const v8bf*)(ga + 8);
    v8bf rb0 = *(const v8bf*)(gb);
    v8bf rb1 = *(const v8bf*)(gb + 8);

    v8f acc[2][4] = {};
    const int nk = K >> 5;

    for (int kt = 0; kt < nk; ++kt) {
        // ---- commit staged registers to LDS ----
        *(v8bf*)&lA[srow * LPAD + skc]     = ra0;
        *(v8bf*)&lA[srow * LPAD + skc + 8] = ra1;
        *(v8bf*)&lB[srow * LPAD + skc]     = rb0;
        *(v8bf*)&lB[srow * LPAD + skc + 8] = rb1;
        __syncthreads();

        // ---- issue next K-step's global loads (waited after the WMMAs) ----
        if (kt + 1 < nk) {
            const bf16_t* na = ga + (size_t)(kt + 1) * 32;
            const bf16_t* nb = gb + (size_t)(kt + 1) * 32;
            ra0 = *(const v8bf*)(na); ra1 = *(const v8bf*)(na + 8);
            rb0 = *(const v8bf*)(nb); rb1 = *(const v8bf*)(nb + 8);
        }

        // ---- assemble fragments (two 16B LDS reads each) ----
        v16bf afrag[2], bfrag[4];
        #pragma unroll
        for (int mt = 0; mt < 2; ++mt) {
            const bf16_t* p = &lA[(wm + mt * 16 + fr) * LPAD + khalf];
            v8bf lo = *(const v8bf*)p;          // K = khalf .. khalf+7
            v8bf hi = *(const v8bf*)(p + 16);   // K = khalf+16 .. khalf+23
            #pragma unroll
            for (int j = 0; j < 8; ++j) { afrag[mt][j] = lo[j]; afrag[mt][j + 8] = hi[j]; }
        }
        #pragma unroll
        for (int nt = 0; nt < 4; ++nt) {
            const bf16_t* p = &lB[(wn + nt * 16 + fr) * LPAD + khalf];
            v8bf lo = *(const v8bf*)p;
            v8bf hi = *(const v8bf*)(p + 16);
            #pragma unroll
            for (int j = 0; j < 8; ++j) { bfrag[nt][j] = lo[j]; bfrag[nt][j + 8] = hi[j]; }
        }

        // ---- 8x v_wmma_f32_16x16x32_bf16 ----
        #pragma unroll
        for (int mt = 0; mt < 2; ++mt)
            #pragma unroll
            for (int nt = 0; nt < 4; ++nt)
                acc[mt][nt] = __builtin_amdgcn_wmma_f32_16x16x32_bf16(
                    /*neg_a=*/false, afrag[mt], /*neg_b=*/false, bfrag[nt],
                    /*c_mod=*/(short)0, acc[mt][nt],
                    /*reuse_a=*/false, /*reuse_b=*/false);
        __syncthreads();
    }

    // ---- epilogue: C layout (VGPR r -> M=r / r+8 by lane half), + bias ----
    const int cm = (lane >> 4) << 3;   // 0 or 8
    const int cn = lane & 15;
    #pragma unroll
    for (int mt = 0; mt < 2; ++mt) {
        #pragma unroll
        for (int nt = 0; nt < 4; ++nt) {
            int col = bn + wn + nt * 16 + cn;
            float b = bias[col];
            #pragma unroll
            for (int r = 0; r < 8; ++r) {
                int row = bm + wm + mt * 16 + cm + r;
                Out[(size_t)row * N + col] = acc[mt][nt][r] + b;
            }
        }
    }
}

// ---------------------------------------------------------------------------
// e1: zr1 epilogue. z1 = 0.875*sigmoid(.) + 0.125 (saved), r1 = sigmoid(.),
// write r1*h2_fl into A's second half for GEMM2.
// ---------------------------------------------------------------------------
__global__ void e1_kernel(const float* __restrict__ P1,
                          const int* __restrict__ hidden,
                          float* __restrict__ z1f,
                          bf16_t* __restrict__ A) {
    int i = blockIdx.x * blockDim.x + threadIdx.x;   // over BATCH*GSZ
    int row = i >> 10, j = i & (GSZ - 1);
    float z = sigmoid_(P1[(size_t)row * HSZ + j]);
    z = MAXF * z + (1.0f - MAXF);
    z1f[i] = z;
    float r = sigmoid_(P1[(size_t)row * HSZ + GSZ + j]);
    float h2fl = (float)hidden[(size_t)row * HSZ + GSZ + j] * INV2P23;
    A[(size_t)row * KDIM + IN_SZ + j] = (bf16_t)(r * h2fl);
}

// ---------------------------------------------------------------------------
// e2: g1 = tanh(.), exact fixed-point h1 update, emit h1 (int32) + h1_fl,
// write h1_fl into A's second half for GEMM3.
// ---------------------------------------------------------------------------
__global__ void e2_kernel(const float* __restrict__ P2,
                          const int* __restrict__ hidden,
                          const float* __restrict__ z1f,
                          int* __restrict__ out_ri,
                          float* __restrict__ out_hf,
                          float* __restrict__ h1fl_buf,
                          bf16_t* __restrict__ A) {
    int i = blockIdx.x * blockDim.x + threadIdx.x;
    int row = i >> 10, j = i & (GSZ - 1);
    float g = tanhf(P2[(size_t)row * GSZ + j]);
    float z = z1f[i];
    long long h1   = (long long)hidden[(size_t)row * HSZ + j];
    long long zfix = (long long)truncf(z * 1024.0f);
    if (zfix < 1) zfix = 1;
    h1 = (h1 * zfix) >> 10;                                 // floor_divide 2^10
    h1 += (long long)truncf((1.0f - z) * g * F2P23);        // trunc fixed-point
    int h1i = (int)h1;
    out_ri[(size_t)row * HSZ + j] = h1i;
    float hf = (float)h1i * INV2P23;
    out_hf[(size_t)row * HSZ + j] = hf;
    h1fl_buf[i] = hf;
    A[(size_t)row * KDIM + IN_SZ + j] = (bf16_t)hf;
}

// ---------------------------------------------------------------------------
// e3: zr2 epilogue; save z2, write r2*h1_fl into A for GEMM4.
// ---------------------------------------------------------------------------
__global__ void e3_kernel(const float* __restrict__ P1,
                          const float* __restrict__ h1fl,
                          float* __restrict__ z2f,
                          bf16_t* __restrict__ A) {
    int i = blockIdx.x * blockDim.x + threadIdx.x;
    int row = i >> 10, j = i & (GSZ - 1);
    float z = sigmoid_(P1[(size_t)row * HSZ + j]);
    z = MAXF * z + (1.0f - MAXF);
    z2f[i] = z;
    float r = sigmoid_(P1[(size_t)row * HSZ + GSZ + j]);
    A[(size_t)row * KDIM + IN_SZ + j] = (bf16_t)(r * h1fl[i]);
}

// ---------------------------------------------------------------------------
// e4: g2 = tanh(.), fixed-point h2 update, emit h2 (int32) + h2_fl.
// ---------------------------------------------------------------------------
__global__ void e4_kernel(const float* __restrict__ P2,
                          const int* __restrict__ hidden,
                          const float* __restrict__ z2f,
                          int* __restrict__ out_ri,
                          float* __restrict__ out_hf) {
    int i = blockIdx.x * blockDim.x + threadIdx.x;
    int row = i >> 10, j = i & (GSZ - 1);
    float g = tanhf(P2[(size_t)row * GSZ + j]);
    float z = z2f[i];
    long long h2   = (long long)hidden[(size_t)row * HSZ + GSZ + j];
    long long zfix = (long long)truncf(z * 1024.0f);
    if (zfix < 1) zfix = 1;
    h2 = (h2 * zfix) >> 10;
    h2 += (long long)truncf((1.0f - z) * g * F2P23);
    int h2i = (int)h2;
    out_ri[(size_t)row * HSZ + GSZ + j] = h2i;
    out_hf[(size_t)row * HSZ + GSZ + j] = (float)h2i * INV2P23;
}

// ---------------------------------------------------------------------------
// optimal_bits = sum(-log2 z1[:, sd:]) + sum(-log2 z2) + 32*sd*BATCH
// ---------------------------------------------------------------------------
__global__ void bits_kernel(const float* __restrict__ z1f,
                            const float* __restrict__ z2f,
                            const int* __restrict__ slice_dim,
                            float* __restrict__ bits) {
    __shared__ float sm[256];
    const float INV_LN2 = 1.44269504088896340736f;
    int gid = blockIdx.x * blockDim.x + threadIdx.x;
    int sd = *slice_dim;
    float v = 0.0f;
    if (gid < BATCH * GSZ) {
        int col = gid & (GSZ - 1);
        if (col >= sd) v = -logf(z1f[gid]) * INV_LN2;
    } else {
        v = -logf(z2f[gid - BATCH * GSZ]) * INV_LN2;
    }
    if (gid == 0) v += 32.0f * (float)sd * (float)BATCH;
    sm[threadIdx.x] = v;
    __syncthreads();
    for (int s = 128; s > 0; s >>= 1) {
        if (threadIdx.x < s) sm[threadIdx.x] += sm[threadIdx.x + s];
        __syncthreads();
    }
    if (threadIdx.x == 0) atomicAdd(bits, sm[0]);
}

// ---------------------------------------------------------------------------
extern "C" void kernel_launch(void* const* d_in, const int* in_sizes, int n_in,
                              void* d_out, int out_size, void* d_ws, size_t ws_size,
                              hipStream_t stream) {
    (void)in_sizes; (void)n_in; (void)out_size; (void)ws_size;

    const float* x      = (const float*)d_in[0];
    const int*   hidden = (const int*)d_in[1];
    const float* W_zr1  = (const float*)d_in[2];
    const float* b_zr1  = (const float*)d_in[3];
    const float* W_g1   = (const float*)d_in[4];
    const float* b_g1   = (const float*)d_in[5];
    const float* W_zr2  = (const float*)d_in[6];
    const float* b_zr2  = (const float*)d_in[7];
    const float* W_g2   = (const float*)d_in[8];
    const float* b_g2   = (const float*)d_in[9];
    const int* slice_dim = (const int*)d_in[10];

    // ---- workspace carve-up (~260 MB) ----
    char* ws = (char*)d_ws;
    bf16_t* A     = (bf16_t*)ws; ws += (size_t)BATCH * KDIM * 2;
    bf16_t* Wzr1t = (bf16_t*)ws; ws += (size_t)KDIM * HSZ * 2;   // [N=2048][K]
    bf16_t* Wg1t  = (bf16_t*)ws; ws += (size_t)KDIM * GSZ * 2;   // [N=1024][K]
    bf16_t* Wzr2t = (bf16_t*)ws; ws += (size_t)KDIM * HSZ * 2;
    bf16_t* Wg2t  = (bf16_t*)ws; ws += (size_t)KDIM * GSZ * 2;
    float*  P1    = (float*)ws;  ws += (size_t)BATCH * HSZ * 4;
    float*  P2    = (float*)ws;  ws += (size_t)BATCH * GSZ * 4;
    float*  z1f   = (float*)ws;  ws += (size_t)BATCH * GSZ * 4;
    float*  z2f   = (float*)ws;  ws += (size_t)BATCH * GSZ * 4;
    float*  h1fl  = (float*)ws;  ws += (size_t)BATCH * GSZ * 4;

    int*   out_ri = (int*)d_out;                                 // recurrent_hidden (int32 bits)
    float* out_hf = (float*)d_out + (size_t)BATCH * HSZ;         // output_hidden
    float* bits   = (float*)d_out + (size_t)2 * BATCH * HSZ;     // optimal_bits

    const dim3 blk(256);
    const int EW = (BATCH * GSZ) / 256;        // elementwise grid (G half)

    // weights -> bf16, transposed to [N][K]
    cvtT_kernel<<<(KDIM * HSZ) / 256, blk, 0, stream>>>(W_zr1, Wzr1t, KDIM, 11);
    cvtT_kernel<<<(KDIM * GSZ) / 256, blk, 0, stream>>>(W_g1,  Wg1t,  KDIM, 10);
    cvtT_kernel<<<(KDIM * HSZ) / 256, blk, 0, stream>>>(W_zr2, Wzr2t, KDIM, 11);
    cvtT_kernel<<<(KDIM * GSZ) / 256, blk, 0, stream>>>(W_g2,  Wg2t,  KDIM, 10);

    // A = [x | h2_fl], zero bits slot
    prep_kernel<<<(BATCH * KDIM) / 256, blk, 0, stream>>>(x, hidden, A, bits);

    // GEMM1: zr1 preact
    gemm_bf16_kernel<<<dim3(HSZ / 128, BATCH / 128), blk, 0, stream>>>(A, Wzr1t, b_zr1, P1, BATCH, HSZ, KDIM);
    e1_kernel<<<EW, blk, 0, stream>>>(P1, hidden, z1f, A);

    // GEMM2: g1 preact
    gemm_bf16_kernel<<<dim3(GSZ / 128, BATCH / 128), blk, 0, stream>>>(A, Wg1t, b_g1, P2, BATCH, GSZ, KDIM);
    e2_kernel<<<EW, blk, 0, stream>>>(P2, hidden, z1f, out_ri, out_hf, h1fl, A);

    // GEMM3: zr2 preact
    gemm_bf16_kernel<<<dim3(HSZ / 128, BATCH / 128), blk, 0, stream>>>(A, Wzr2t, b_zr2, P1, BATCH, HSZ, KDIM);
    e3_kernel<<<EW, blk, 0, stream>>>(P1, h1fl, z2f, A);

    // GEMM4: g2 preact
    gemm_bf16_kernel<<<dim3(GSZ / 128, BATCH / 128), blk, 0, stream>>>(A, Wg2t, b_g2, P2, BATCH, GSZ, KDIM);
    e4_kernel<<<EW, blk, 0, stream>>>(P2, hidden, z2f, out_ri, out_hf);

    // optimal_bits reduction
    bits_kernel<<<(2 * BATCH * GSZ) / 256, blk, 0, stream>>>(z1f, z2f, slice_dim, bits);
}
